// AttentionHeadAdaptive_5523327943002
// MI455X (gfx1250) — matrix operand order repaired and checked by reading, not compiled, not run on hardware
//
#include <hip/hip_runtime.h>
#include <hip/hip_bf16.h>
#include <stdint.h>

// Problem constants (match reference)
#define Bn   16
#define Ln   196
#define Dn   512
#define Mn   4096
#define NROWS 4097
#define SSTRIDE 4104            // padded per-batch score stride (floats)
#define ZSEG 8

// d_ws layout (bytes)
#define WS_SCORES 0
#define WS_WVT    (Bn * SSTRIDE * 4)                  // 262656
#define WS_WUT    (WS_WVT + Dn * (Dn/2) * 4)          // +524288
#define WS_IDX    (WS_WUT + Dn * (Dn/2) * 4)          // +524288
#define WS_ZPART  (WS_IDX + 256)                      // 16*8*512 floats

// d_out layout (float elements)
#define OUT_Z     0
#define OUT_FEAT  (Bn * Dn)                                  // 8192
#define OUT_FREQ  (OUT_FEAT + (size_t)Bn * Mn * Dn)          // 33562624
#define OUT_MIN   (OUT_FREQ + (size_t)Bn * Mn)               // 33628160

// GEMM tiling: 32 rows per tile, 4 tiles per persistent WG
#define TILES_PER_WG 4
#define LDSF_BYTES   (32 * Dn * 4)          // 65536: f32 staging tile
#define LDSA_BYTES   (32 * 260 * 4)         // 33280: bf16 tile (row stride 260 dw)
#define LDSP_BYTES   (8 * 32 * 4)           // 1024 : per-wave score partials
#define SMEM_BYTES   (LDSF_BYTES + LDSA_BYTES + LDSP_BYTES)

typedef __attribute__((ext_vector_type(16))) __bf16 v16bf;
typedef __attribute__((ext_vector_type(8)))  float  v8f;

union FragAB { v16bf v; uint32_t u[8]; uint4 q[2]; };

// ---- CDNA5 async global->LDS path (guarded; falls back to sync loads) ----
#if defined(__has_builtin)
#if __has_builtin(__builtin_amdgcn_global_load_async_to_lds_b128)
#define HAS_ASYNC 1
#endif
#endif

#if defined(HAS_ASYNC)
// Builtin prototype (from the clang diagnostic): first param is
// 'int __attribute__((vector_size(16)))*' in addrspace(1) (printed
// '__device__'); LDS side is addrspace(3). Cast via integer to reach the
// exact pointer types (generic shared addr low 32 bits == LDS offset).
typedef int v4i_vs __attribute__((vector_size(16)));
typedef __attribute__((address_space(1))) v4i_vs* g_v4i_p;
typedef __attribute__((address_space(3))) v4i_vs* l_v4i_p;
#define ASYNC_CP_B128(gp, lp)                                                  \
    __builtin_amdgcn_global_load_async_to_lds_b128(                            \
        (g_v4i_p)(uintptr_t)(gp),                                              \
        (l_v4i_p)(unsigned)(uintptr_t)(lp), 0, 0)
#if __has_builtin(__builtin_amdgcn_s_wait_asynccnt)
#define WAIT_ASYNC0() __builtin_amdgcn_s_wait_asynccnt(0)
#else
#define WAIT_ASYNC0() asm volatile("s_wait_asynccnt 0" ::: "memory")
#endif
#endif

__device__ __forceinline__ uint32_t pk_bf16(float a, float b) {
    unsigned short lo = __builtin_bit_cast(unsigned short, (__bf16)a);
    unsigned short hi = __builtin_bit_cast(unsigned short, (__bf16)b);
    return (uint32_t)lo | ((uint32_t)hi << 16);
}
__device__ __forceinline__ float sigmoidf(float x) { return 1.0f / (1.0f + __expf(-x)); }

// ---------------------------------------------------------------------------
// Kernel 1: transpose + convert Wv/Wu (512x512 f32 [k][n]) to packed bf16
// Wt[n][kpair] so WMMA B fragments are two contiguous b128 loads per lane.
// ---------------------------------------------------------------------------
__global__ __launch_bounds__(128) void k_prep_w(const float* __restrict__ Wv,
                                                const float* __restrict__ Wu,
                                                uint32_t* __restrict__ Wvt,
                                                uint32_t* __restrict__ Wut) {
    int n  = blockIdx.x;
    int kp = blockIdx.y * blockDim.x + threadIdx.x;   // 0..255
    float a = Wv[(2 * kp) * Dn + n];
    float b = Wv[(2 * kp + 1) * Dn + n];
    Wvt[n * 256 + kp] = pk_bf16(a, b);
    a = Wu[(2 * kp) * Dn + n];
    b = Wu[(2 * kp + 1) * Dn + n];
    Wut[n * 256 + kp] = pk_bf16(a, b);
}

// ---------------------------------------------------------------------------
// A-tile staging helpers (32 rows x 512 f32).
// Async: wave w streams rows 4w..4w+3 (contiguous 8 KB) to LDS, 16 x b128.
// ---------------------------------------------------------------------------
__device__ __forceinline__ void tile_prefetch(const float* gRowBase, char* ldsF,
                                              int w, int lane) {
#if defined(HAS_ASYNC)
    const char* g = (const char*)(gRowBase + (size_t)(4 * w) * Dn) + lane * 16;
    char* l = ldsF + (size_t)(4 * w) * Dn * 4 + lane * 16;
    #pragma unroll
    for (int k = 0; k < 16; ++k)
        ASYNC_CP_B128(g + k * 512, l + k * 512);
#else
    (void)gRowBase; (void)ldsF; (void)w; (void)lane;
#endif
}

__device__ __forceinline__ void tile_finalize(const float* gRowBase,
                                              const char* ldsF,
                                              uint32_t* ldsA, int tid) {
    const int r = tid >> 3;              // 0..31
    const int c = (tid & 7) * 64;        // 64 floats per thread
    uint32_t* dst = ldsA + r * 260 + (c >> 1);
#if defined(HAS_ASYNC)
    WAIT_ASYNC0();
    __syncthreads();                      // all waves' async stores landed
    const float4* src = (const float4*)((const float*)ldsF + (size_t)r * Dn + c);
#else
    const float4* src = (const float4*)(gRowBase + (size_t)r * Dn + c);
#endif
    #pragma unroll
    for (int i = 0; i < 16; ++i) {
        float4 f = src[i];
        dst[2 * i]     = pk_bf16(f.x, f.y);
        dst[2 * i + 1] = pk_bf16(f.z, f.w);
    }
    __syncthreads();
}

// ---------------------------------------------------------------------------
// Kernel 2: persistent bf16 WMMA GEMM + fused epilogue -> attn scores.
// grid (32, 16), block 256 (8 waves), dynamic LDS = SMEM_BYTES.
// Each WG: 4 tiles of 32 rows x 512 cols; per k-step each wave loads its
// 4 N-tile B fragments ONCE and reuses them across 2 M-tiles (halves L2
// weight traffic vs 16-row tiles). Next A tile streams in via async-to-LDS
// while WMMAs of the current tile execute.
// ---------------------------------------------------------------------------
__global__ __launch_bounds__(256) void k_gemm_scores(
    const float* __restrict__ feat_mem,
    const uint32_t* __restrict__ Wvt, const uint32_t* __restrict__ Wut,
    const float* __restrict__ bv, const float* __restrict__ bu,
    const float* __restrict__ Ww, const float* __restrict__ bw,
    float* __restrict__ scores)
{
    extern __shared__ char smem[];
    char*     ldsF     = smem;                                  // f32 staging
    uint32_t* ldsA     = (uint32_t*)(smem + LDSF_BYTES);        // bf16 tile
    float*    lds_part = (float*)(smem + LDSF_BYTES + LDSA_BYTES);

    const int tid = threadIdx.x;
    const int b = blockIdx.y;
    const int lane = tid & 31, w = tid >> 5;
    const int m16 = lane & 15, khalf = lane >> 4;

    const int t0 = blockIdx.x * TILES_PER_WG;                   // 32-row tiles
    const float* gbase0 = feat_mem + ((size_t)b * Mn + (size_t)t0 * 32) * Dn;

    tile_prefetch(gbase0, ldsF, w, lane);

    for (int it = 0; it < TILES_PER_WG; ++it) {
        const float* gbase = gbase0 + (size_t)it * 32 * Dn;
        tile_finalize(gbase, ldsF, ldsA, tid);
        if (it + 1 < TILES_PER_WG)
            tile_prefetch(gbase + (size_t)32 * Dn, ldsF, w, lane);

        const v8f zero8 = {0.f, 0.f, 0.f, 0.f, 0.f, 0.f, 0.f, 0.f};
        v8f accV[2][4] = {{zero8, zero8, zero8, zero8}, {zero8, zero8, zero8, zero8}};
        v8f accU[2][4] = {{zero8, zero8, zero8, zero8}, {zero8, zero8, zero8, zero8}};

        const uint32_t* aRow0 = ldsA + m16 * 260;
        const uint32_t* aRow1 = ldsA + (16 + m16) * 260;

        for (int ks = 0; ks < 16; ++ks) {
            FragAB a0, a1;
            {
                const uint4* ap = (const uint4*)(aRow0 + ks * 16 + khalf * 8);
                a0.q[0] = ap[0]; a0.q[1] = ap[1];
            }
            {
                const uint4* ap = (const uint4*)(aRow1 + ks * 16 + khalf * 8);
                a1.q[0] = ap[0]; a1.q[1] = ap[1];
            }
            #pragma unroll
            for (int j = 0; j < 4; ++j) {
                const int n = (w * 4 + j) * 16 + m16;
                FragAB bf;
                const uint4* bp = (const uint4*)(Wvt + n * 256 + ks * 16 + khalf * 8);
                bf.q[0] = bp[0]; bf.q[1] = bp[1];
                accV[0][j] = __builtin_amdgcn_wmma_f32_16x16x32_bf16(
                    false, a0.v, false, bf.v, (short)0, accV[0][j], false, false);
                accV[1][j] = __builtin_amdgcn_wmma_f32_16x16x32_bf16(
                    false, a1.v, false, bf.v, (short)0, accV[1][j], false, false);
                const uint4* up = (const uint4*)(Wut + n * 256 + ks * 16 + khalf * 8);
                bf.q[0] = up[0]; bf.q[1] = up[1];
                accU[0][j] = __builtin_amdgcn_wmma_f32_16x16x32_bf16(
                    false, a0.v, false, bf.v, (short)0, accU[0][j], false, false);
                accU[1][j] = __builtin_amdgcn_wmma_f32_16x16x32_bf16(
                    false, a1.v, false, bf.v, (short)0, accU[1][j], false, false);
            }
        }

        // Fused epilogue: tanh(V)*sigmoid(U)*Ww[n], reduce over N.
        float rp[2][8] = {{0,0,0,0,0,0,0,0},{0,0,0,0,0,0,0,0}};
        #pragma unroll
        for (int j = 0; j < 4; ++j) {
            const int n = (w * 4 + j) * 16 + m16;
            const float bvn = bv[n], bun = bu[n], wwn = Ww[n];
            #pragma unroll
            for (int mt = 0; mt < 2; ++mt) {
                #pragma unroll
                for (int v = 0; v < 8; ++v) {
                    float V = tanhf(accV[mt][j][v] + bvn);
                    float U = sigmoidf(accU[mt][j][v] + bun);
                    rp[mt][v] += V * U * wwn;
                }
            }
        }
        #pragma unroll
        for (int mt = 0; mt < 2; ++mt) {
            #pragma unroll
            for (int v = 0; v < 8; ++v) {
                float r = rp[mt][v];
                r += __shfl_xor(r, 1, 16);
                r += __shfl_xor(r, 2, 16);
                r += __shfl_xor(r, 4, 16);
                r += __shfl_xor(r, 8, 16);
                if (m16 == 0)
                    lds_part[w * 32 + mt * 16 + v + khalf * 8] = r;
            }
        }
        __syncthreads();
        if (tid < 32) {
            float s = 0.f;
            #pragma unroll
            for (int ww = 0; ww < 8; ++ww) s += lds_part[ww * 32 + tid];
            scores[b * SSTRIDE + (t0 + it) * 32 + tid] = sigmoidf(s + bw[0]);
        }
        __syncthreads();   // protect ldsA / lds_part before next tile
    }
}

// ---------------------------------------------------------------------------
// Kernel 3: cls row (row 4096) score per batch. grid 16, block 256.
// ---------------------------------------------------------------------------
__global__ __launch_bounds__(256) void k_cls(
    const float* __restrict__ x,
    const float* __restrict__ Wv, const float* __restrict__ bv,
    const float* __restrict__ Wu, const float* __restrict__ bu,
    const float* __restrict__ Ww, const float* __restrict__ bw,
    float* __restrict__ scores)
{
    __shared__ float xs[Dn];
    __shared__ float red[256];
    const int b = blockIdx.x, tid = threadIdx.x;
    const float* xr = x + (size_t)b * Ln * Dn;   // x[b, 0, :]
    xs[tid] = xr[tid];
    xs[tid + 256] = xr[tid + 256];
    __syncthreads();
    const int d0 = tid, d1 = tid + 256;
    float v0 = 0.f, v1 = 0.f, u0 = 0.f, u1 = 0.f;
    for (int k = 0; k < Dn; ++k) {
        float xv = xs[k];
        const float* wr = Wv + k * Dn;
        const float* ur = Wu + k * Dn;
        v0 += xv * wr[d0]; v1 += xv * wr[d1];
        u0 += xv * ur[d0]; u1 += xv * ur[d1];
    }
    float p0 = tanhf(v0 + bv[d0]) * sigmoidf(u0 + bu[d0]);
    float p1 = tanhf(v1 + bv[d1]) * sigmoidf(u1 + bu[d1]);
    red[tid] = p0 * Ww[d0] + p1 * Ww[d1];
    __syncthreads();
    for (int s = 128; s > 0; s >>= 1) {
        if (tid < s) red[tid] += red[tid + s];
        __syncthreads();
    }
    if (tid == 0) scores[b * SSTRIDE + Mn] = sigmoidf(red[0] + bw[0]);
}

// ---------------------------------------------------------------------------
// Kernel 4a: z partials over 512-row segments (deterministic), with
// global_prefetch on the streamed column. grid (4, 16, 8), block 128.
// ---------------------------------------------------------------------------
__global__ __launch_bounds__(128) void k_zpart(
    const float* __restrict__ feat_mem, const float* __restrict__ x,
    const float* __restrict__ scores, float* __restrict__ zpart)
{
    const int b = blockIdx.y, seg = blockIdx.z;
    const int d = blockIdx.x * 128 + threadIdx.x;
    const float* col = feat_mem + (size_t)b * Mn * Dn + d;
    const float* sr = scores + b * SSTRIDE;
    const int i0 = seg * (Mn / ZSEG);
    float acc = 0.f;
    for (int i = i0; i < i0 + Mn / ZSEG; ++i) {
        if ((d & 15) == 0)
            __builtin_prefetch(col + (size_t)(i + 16) * Dn, 0, 3);
        acc += sr[i] * col[(size_t)i * Dn];
    }
    if (seg == ZSEG - 1)
        acc += sr[Mn] * x[(size_t)b * Ln * Dn + d];
    zpart[((size_t)b * ZSEG + seg) * Dn + d] = acc;
}

// Kernel 4b: fixed-order combine. grid (4, 16), block 128.
__global__ __launch_bounds__(128) void k_zsum(const float* __restrict__ zpart,
                                              float* __restrict__ out)
{
    const int b = blockIdx.y;
    const int d = blockIdx.x * 128 + threadIdx.x;
    float s = 0.f;
    #pragma unroll
    for (int seg = 0; seg < ZSEG; ++seg)
        s += zpart[((size_t)b * ZSEG + seg) * Dn + d];
    out[OUT_Z + b * Dn + d] = s;
}

// ---------------------------------------------------------------------------
// Kernel 5: argmin(score) and argmax(ratio) per batch (first-occurrence ties).
// grid 16, block 256.
// ---------------------------------------------------------------------------
__global__ __launch_bounds__(256) void k_idx(
    const float* __restrict__ scores, const int* __restrict__ freq_mem,
    const int* __restrict__ min_mem, int* __restrict__ idxbuf)
{
    __shared__ float sv[256];
    __shared__ int   si[256];
    __shared__ int   s_attn;
    const int b = blockIdx.x, tid = threadIdx.x;
    const float* sr = scores + b * SSTRIDE;

    float best = __builtin_inff(); int bi = 0x7fffffff;
    for (int i = tid; i < NROWS; i += 256) {
        float v = sr[i];
        if (v < best) { best = v; bi = i; }
    }
    sv[tid] = best; si[tid] = bi;
    __syncthreads();
    for (int s = 128; s > 0; s >>= 1) {
        if (tid < s) {
            float v2 = sv[tid + s]; int i2 = si[tid + s];
            if (v2 < sv[tid] || (v2 == sv[tid] && i2 < si[tid])) { sv[tid] = v2; si[tid] = i2; }
        }
        __syncthreads();
    }
    if (tid == 0) s_attn = si[0];
    __syncthreads();
    const int attn = s_attn;

    float bestr = -1.0f; int bri = 0x7fffffff;
    for (int i = tid; i < NROWS; i += 256) {
        int fi = (i < Mn ? freq_mem[b * Mn + i] : 0) + 1;
        int mi = (i < Mn ? min_mem[b * Mn + i] : 0) + (i == attn ? 1 : 0);
        float ratio = (fi > 5) ? (float)mi / (float)fi : 0.0f;
        if (ratio > bestr) { bestr = ratio; bri = i; }
    }
    sv[tid] = bestr; si[tid] = bri;
    __syncthreads();
    for (int s = 128; s > 0; s >>= 1) {
        if (tid < s) {
            float v2 = sv[tid + s]; int i2 = si[tid + s];
            if (v2 > sv[tid] || (v2 == sv[tid] && i2 < si[tid])) { sv[tid] = v2; si[tid] = i2; }
        }
        __syncthreads();
    }
    if (tid == 0) { idxbuf[b * 2] = attn; idxbuf[b * 2 + 1] = si[0]; }
}

// ---------------------------------------------------------------------------
// Kernel 6: row-removal gather -> feat_out / freq_out / min_out.
// grid (4096, 16), block 128 (float4 per thread).
// ---------------------------------------------------------------------------
__global__ __launch_bounds__(128) void k_gather(
    const float* __restrict__ feat_mem, const float* __restrict__ x,
    const int* __restrict__ freq_mem, const int* __restrict__ min_mem,
    const int* __restrict__ idxbuf, float* __restrict__ out)
{
    const int i = blockIdx.x, b = blockIdx.y, tid = threadIdx.x;
    const int attn = idxbuf[b * 2];
    const int rm   = idxbuf[b * 2 + 1];
    const int src  = i + (i >= rm ? 1 : 0);
    const float* srow = (src == Mn) ? (x + (size_t)b * Ln * Dn)
                                    : (feat_mem + ((size_t)b * Mn + src) * Dn);
    const float4* s4 = (const float4*)srow;
    float4* drow = (float4*)(out + OUT_FEAT + ((size_t)b * Mn + i) * Dn);
    drow[tid] = s4[tid];
    if (tid == 0) {
        int fi = (src < Mn ? freq_mem[b * Mn + src] : 0) + 1;
        int mi = (src < Mn ? min_mem[b * Mn + src] : 0) + (src == attn ? 1 : 0);
        ((int*)(out + OUT_FREQ))[b * Mn + i] = fi;
        ((int*)(out + OUT_MIN))[b * Mn + i]  = mi;
    }
}

// ---------------------------------------------------------------------------
extern "C" void kernel_launch(void* const* d_in, const int* in_sizes, int n_in,
                              void* d_out, int out_size, void* d_ws, size_t ws_size,
                              hipStream_t stream)
{
    const float* x        = (const float*)d_in[0];
    const float* feat_mem = (const float*)d_in[1];
    const int*   freq_mem = (const int*)d_in[2];
    const int*   min_mem  = (const int*)d_in[3];
    const float* Wv       = (const float*)d_in[4];
    const float* bv       = (const float*)d_in[5];
    const float* Wu       = (const float*)d_in[6];
    const float* bu       = (const float*)d_in[7];
    const float* Ww       = (const float*)d_in[8];
    const float* bw       = (const float*)d_in[9];
    float* out = (float*)d_out;

    char* ws = (char*)d_ws;
    float*    scores = (float*)(ws + WS_SCORES);
    uint32_t* Wvt    = (uint32_t*)(ws + WS_WVT);
    uint32_t* Wut    = (uint32_t*)(ws + WS_WUT);
    int*      idxbuf = (int*)(ws + WS_IDX);
    float*    zpart  = (float*)(ws + WS_ZPART);

    k_prep_w<<<dim3(512, 2), 128, 0, stream>>>(Wv, Wu, Wvt, Wut);
    k_gemm_scores<<<dim3(128 / TILES_PER_WG, Bn), 256, SMEM_BYTES, stream>>>(
        feat_mem, Wvt, Wut, bv, bu, Ww, bw, scores);
    k_cls<<<Bn, 256, 0, stream>>>(x, Wv, bv, Wu, bu, Ww, bw, scores);
    k_zpart<<<dim3(4, Bn, ZSEG), 128, 0, stream>>>(feat_mem, x, scores, zpart);
    k_zsum<<<dim3(4, Bn), 128, 0, stream>>>(zpart, out);
    k_idx<<<Bn, 256, 0, stream>>>(scores, freq_mem, min_mem, idxbuf);
    k_gather<<<dim3(Mn, Bn), 128, 0, stream>>>(feat_mem, x, freq_mem, min_mem,
                                               idxbuf, out);
}
